// AxialTransformerLayer_58600533786748
// MI455X (gfx1250) — matrix operand order repaired and checked by reading, not compiled, not run on hardware
//
#include <hip/hip_runtime.h>
#include <math.h>

// ---------- problem constants (fixed by the reference) ----------
#define R_  256
#define C_  256
#define D_  768
#define H_  12
#define HD_ 64
#define F_  3072
#define M_  (R_ * C_)     // 65536 token rows

// ---------- CDNA5 WMMA vector types ----------
typedef __attribute__((ext_vector_type(16))) __bf16 v16bf;
typedef __attribute__((ext_vector_type(8)))  float  v8f;

// ---------- gfx1250 async global->LDS path (probe at compile time) ----------
#if defined(__gfx1250__) && __has_builtin(__builtin_amdgcn_global_load_async_to_lds_b128)
#define USE_ASYNC_LDS 1
#define AS1 __attribute__((address_space(1)))
#define AS3 __attribute__((address_space(3)))
typedef int v4i_ __attribute__((vector_size(16)));   // matches builtin's pointee type
#if __has_builtin(__builtin_amdgcn_s_wait_asynccnt)
#define WAIT_ASYNC(n) __builtin_amdgcn_s_wait_asynccnt(n)
#else
#define WAIT_ASYNC(n) asm volatile("s_wait_asynccnt %0" ::"i"(n) : "memory")
#endif
__device__ __forceinline__ void async_ld_b128(void* g, void* l) {
  __builtin_amdgcn_global_load_async_to_lds_b128(
      (AS1 v4i_*)(AS1 void*)g, (AS3 v4i_*)(AS3 void*)l, 0, 0);
}
#else
#define USE_ASYNC_LDS 0
#endif

__device__ __forceinline__ unsigned short f2bf(float x) {
  union { float f; unsigned int u; } a; a.f = x;
  unsigned int r = a.u + 0x7FFFu + ((a.u >> 16) & 1u);  // round-to-nearest-even
  return (unsigned short)(r >> 16);
}

enum {
  MODE_PROJ_Q = 0,   // bias, *scale, zero padded query rows, scatter to q_t[h][i][r][d]
  MODE_PROJ_K,       // bias, scatter to k_t[h][j][r][d]
  MODE_PROJ_V,       // bias, scatter to v_t[h][r][d][j]
  MODE_SCORES,       // per-head: scores[h][i][j] f32
  MODE_CTX,          // per-head: ctx_std[(r*256+i)*768 + h*64+d] bf16
  MODE_OUTPROJ,      // bias + residual -> f32 (optionally transposed row map)
  MODE_FFN1,         // bias + exact GELU -> bf16 [M,3072]
  MODE_FFN2          // bias + residual -> f32 [M,768]
};

union FragU { uint4 u4[2]; v16bf v; };

// ---------------------------------------------------------------------------
// Generic bf16 WMMA GEMM:  C[M,N] = A[M,K] (row-major) * B[N,K] (row-major)^T
// Block tile 128x64, BK=64, 256 threads = 8 waves (wave32), each wave 32x32.
// Double-buffered LDS; global->LDS via async DMA (ASYNCcnt) when available.
// ---------------------------------------------------------------------------
template <int MODE>
__global__ __launch_bounds__(256) void gemm_wmma(
    const unsigned short* __restrict__ A,
    const unsigned short* __restrict__ B,
    const float*          __restrict__ bias,
    const float*          __restrict__ residual,
    float*                __restrict__ outF,
    unsigned short*       __restrict__ outB,
    const unsigned char*  __restrict__ pm,
    int K, int lda, int ldb, float scale, int flagT)
{
  constexpr int BM = 128, BN = 64, BK = 64;
  constexpr int LSA = BK + 8, LSB = BK + 8;   // 144-byte LDS row stride: 16B aligned, bank-spread
  __shared__ unsigned short As[2 * BM * LSA];
  __shared__ unsigned short Bs[2 * BN * LSB];

  const int tid  = threadIdx.x;
  const int lane = tid & 31;
  const int wave = tid >> 5;
  const int wm   = wave >> 1;           // 0..3
  const int wn   = wave & 1;            // 0..1
  const int bm   = blockIdx.y * BM;
  const int bn   = blockIdx.x * BN;
  const int z    = blockIdx.z;          // head index for batched modes

  if (MODE == MODE_SCORES) {
    A    += (size_t)z * 256 * 16384;
    B    += (size_t)z * 256 * 16384;
    outF += (size_t)z * 65536;
  } else if (MODE == MODE_CTX) {
    A += (size_t)z * 65536;             // probs[h]
    B += (size_t)z * 16384 * 256;       // v_t[h]
  }

  // issue one K-tile (kb) into LDS buffer `buf` (A: 4 x b128, B: 2 x b128 per thread)
  auto issueTile = [&](int kb, int buf) {
    const int k0 = kb * BK;
    unsigned short* as = &As[buf * (BM * LSA)];
    unsigned short* bs = &Bs[buf * (BN * LSB)];
#if USE_ASYNC_LDS
    #pragma unroll
    for (int t = 0; t < 4; ++t) {
      int u  = tid + t * 256;
      int rw = u >> 3;
      int c8 = (u & 7) << 3;
      async_ld_b128(const_cast<unsigned short*>(&A[(size_t)(bm + rw) * lda + k0 + c8]),
                    &as[rw * LSA + c8]);
    }
    #pragma unroll
    for (int t = 0; t < 2; ++t) {
      int u  = tid + t * 256;
      int rw = u >> 3;
      int c8 = (u & 7) << 3;
      async_ld_b128(const_cast<unsigned short*>(&B[(size_t)(bn + rw) * ldb + k0 + c8]),
                    &bs[rw * LSB + c8]);
    }
#else
    // fallback: batch all 6 global loads into distinct regs, then store to LDS,
    // so the loads pipeline instead of load->wait->store one at a time.
    uint4 ra[4], rb[2];
    #pragma unroll
    for (int t = 0; t < 4; ++t) {
      int u = tid + t * 256, rw = u >> 3, c8 = (u & 7) << 3;
      ra[t] = *(const uint4*)&A[(size_t)(bm + rw) * lda + k0 + c8];
    }
    #pragma unroll
    for (int t = 0; t < 2; ++t) {
      int u = tid + t * 256, rw = u >> 3, c8 = (u & 7) << 3;
      rb[t] = *(const uint4*)&B[(size_t)(bn + rw) * ldb + k0 + c8];
    }
    #pragma unroll
    for (int t = 0; t < 4; ++t) {
      int u = tid + t * 256, rw = u >> 3, c8 = (u & 7) << 3;
      *(uint4*)&as[rw * LSA + c8] = ra[t];
    }
    #pragma unroll
    for (int t = 0; t < 2; ++t) {
      int u = tid + t * 256, rw = u >> 3, c8 = (u & 7) << 3;
      *(uint4*)&bs[rw * LSB + c8] = rb[t];
    }
#endif
  };

  v8f acc[2][2];
  #pragma unroll
  for (int a = 0; a < 2; ++a)
    #pragma unroll
    for (int b = 0; b < 2; ++b)
      #pragma unroll
      for (int e = 0; e < 8; ++e) acc[a][b][e] = 0.0f;

  const int kBlocks = K / BK;
  int buf = 0;
  issueTile(0, 0);

  for (int kb = 0; kb < kBlocks; ++kb) {
    if (kb + 1 < kBlocks) {
      issueTile(kb + 1, buf ^ 1);       // prefetch next tile into other buffer
#if USE_ASYNC_LDS
      WAIT_ASYNC(6);                    // 6 newer copies may remain outstanding
#endif
    } else {
#if USE_ASYNC_LDS
      WAIT_ASYNC(0);
#endif
    }
    __syncthreads();                    // tile kb visible to all waves

    const unsigned short* as = &As[buf * (BM * LSA)];
    const unsigned short* bs = &Bs[buf * (BN * LSB)];
    #pragma unroll
    for (int ks = 0; ks < 2; ++ks) {          // two k=32 WMMA steps per stage
      const int kk = ks * 32;
      FragU fa[2], fb[2];
      const int half8  = (lane >> 4) << 3;    // A lane K-split: {0..7,16..23}/{8..15,24..31}
      const int half16 = (lane >> 4) << 3;    // B lane K-split: {0..7,16..23}/{8..15,24..31}
      #pragma unroll
      for (int tm = 0; tm < 2; ++tm) {
        const int rowA = wm * 32 + tm * 16 + (lane & 15);
        fa[tm].u4[0] = *(const uint4*)&as[rowA * LSA + kk + half8];
        fa[tm].u4[1] = *(const uint4*)&as[rowA * LSA + kk + half8 + 16];
      }
      #pragma unroll
      for (int tn = 0; tn < 2; ++tn) {
        const int rowB = wn * 32 + tn * 16 + (lane & 15);
        fb[tn].u4[0] = *(const uint4*)&bs[rowB * LSB + kk + half16];
        fb[tn].u4[1] = *(const uint4*)&bs[rowB * LSB + kk + half16 + 16];
      }
      #pragma unroll
      for (int tm = 0; tm < 2; ++tm)
        #pragma unroll
        for (int tn = 0; tn < 2; ++tn)
          acc[tm][tn] = __builtin_amdgcn_wmma_f32_16x16x32_bf16(
              false, fa[tm].v, false, fb[tn].v, (short)0, acc[tm][tn],
              false, false);
    }
    __syncthreads();                    // all waves done reading buf before reuse
    buf ^= 1;
  }

  // ------------------------- epilogue -------------------------
  #pragma unroll
  for (int tm = 0; tm < 2; ++tm) {
    #pragma unroll
    for (int tn = 0; tn < 2; ++tn) {
      #pragma unroll
      for (int e = 0; e < 8; ++e) {
        const int gm = bm + wm * 32 + tm * 16 + ((lane >> 4) << 3) + e;
        const int gn = bn + wn * 32 + tn * 16 + (lane & 15);
        float v = acc[tm][tn][e];
        if (MODE == MODE_PROJ_Q) {
          v = (v + bias[gn]) * scale;
          const int mi = flagT ? (((gm & 255) << 8) | (gm >> 8)) : gm;
          if (pm[mi]) v = 0.0f;                       // zero padded query rows
          const int h = gn >> 6, d = gn & 63, i = gm & 255, r = gm >> 8;
          outB[((((h << 8) | i) << 8) | r) * 64 + d] = f2bf(v);
        } else if (MODE == MODE_PROJ_K) {
          v += bias[gn];
          const int h = gn >> 6, d = gn & 63, j = gm & 255, r = gm >> 8;
          outB[((((h << 8) | j) << 8) | r) * 64 + d] = f2bf(v);
        } else if (MODE == MODE_PROJ_V) {
          v += bias[gn];
          const int h = gn >> 6, d = gn & 63, j = gm & 255, r = gm >> 8;
          outB[((((h << 8) | r) * 64 + d) << 8) | j] = f2bf(v);
        } else if (MODE == MODE_SCORES) {
          outF[(gm << 8) | gn] = v;
        } else if (MODE == MODE_CTX) {
          const int r = gn >> 6, d = gn & 63;        // gm = query index i
          outB[(size_t)((r << 8) | gm) * D_ + (z << 6) + d] = f2bf(v);
        } else if (MODE == MODE_OUTPROJ) {
          v += bias[gn];
          const int orow = flagT ? (((gm & 255) << 8) | (gm >> 8)) : gm;
          const size_t oi = (size_t)orow * D_ + gn;
          outF[oi] = residual[oi] + v;
        } else if (MODE == MODE_FFN1) {
          v += bias[gn];
          v = 0.5f * v * (1.0f + erff(v * 0.70710678118654752f));  // exact GELU
          outB[(size_t)gm * F_ + gn] = f2bf(v);
        } else {  // MODE_FFN2
          v += bias[gn];
          const size_t oi = (size_t)gm * D_ + gn;
          outF[oi] = residual[oi] + v;
        }
      }
    }
  }
}

// ---------------------------------------------------------------------------
// LayerNorm over D=768, one row per block; optional transposed bf16 output.
// ---------------------------------------------------------------------------
__global__ __launch_bounds__(256) void ln_kernel(
    const float* __restrict__ x, const float* __restrict__ g,
    const float* __restrict__ b, unsigned short* __restrict__ y, int flagT)
{
  __shared__ float s1[256], s2[256];
  const int m = blockIdx.x;
  const int t = threadIdx.x;
  const float* row = x + (size_t)m * D_;
  const float v0 = row[t], v1 = row[t + 256], v2 = row[t + 512];
  s1[t] = v0 + v1 + v2;
  s2[t] = v0 * v0 + v1 * v1 + v2 * v2;
  __syncthreads();
  for (int s = 128; s > 0; s >>= 1) {
    if (t < s) { s1[t] += s1[t + s]; s2[t] += s2[t + s]; }
    __syncthreads();
  }
  const float mu   = s1[0] * (1.0f / D_);
  const float var  = s2[0] * (1.0f / D_) - mu * mu;
  const float rstd = rsqrtf(var + 1e-12f);
  const int om = flagT ? (((m & 255) << 8) | (m >> 8)) : m;
  unsigned short* yr = y + (size_t)om * D_;
  yr[t]       = f2bf((v0 - mu) * rstd * g[t]       + b[t]);
  yr[t + 256] = f2bf((v1 - mu) * rstd * g[t + 256] + b[t + 256]);
  yr[t + 512] = f2bf((v2 - mu) * rstd * g[t + 512] + b[t + 512]);
}

// ---------------------------------------------------------------------------
// Masked softmax over j=256; one (h,i) row per block; bf16 probs out.
// ---------------------------------------------------------------------------
__global__ __launch_bounds__(256) void softmax_kernel(
    const float* __restrict__ scores, unsigned short* __restrict__ probs,
    const unsigned char* __restrict__ pm, int flagT)
{
  __shared__ float sh[256];
  const int hi = blockIdx.x;       // h*256 + i
  const int j  = threadIdx.x;
  float s = scores[(size_t)hi * 256 + j];
  const int mj = flagT ? (j << 8) : j;   // key mask: pmask[0,0,j] (row) / pmask[0,j,0] (col)
  if (pm[mj]) s = -10000.0f;
  sh[j] = s; __syncthreads();
  for (int k = 128; k > 0; k >>= 1) { if (j < k) sh[j] = fmaxf(sh[j], sh[j + k]); __syncthreads(); }
  const float mx = sh[0]; __syncthreads();
  const float e = __expf(s - mx);
  sh[j] = e; __syncthreads();
  for (int k = 128; k > 0; k >>= 1) { if (j < k) sh[j] += sh[j + k]; __syncthreads(); }
  probs[(size_t)hi * 256 + j] = f2bf(e / sh[0]);
}

__global__ void cvt_bf16(const float* __restrict__ src,
                         unsigned short* __restrict__ dst, int n)
{
  const int i = blockIdx.x * 256 + threadIdx.x;
  if (i < n) dst[i] = f2bf(src[i]);
}

// ---------------------------------------------------------------------------
extern "C" void kernel_launch(void* const* d_in, const int* in_sizes, int n_in,
                              void* d_out, int out_size, void* d_ws, size_t ws_size,
                              hipStream_t stream)
{
  (void)in_sizes; (void)out_size; (void)ws_size;
  const float*         x_in = (const float*)d_in[0];
  const unsigned char* pm   = (const unsigned char*)d_in[1];
  const float* p[26] = {nullptr};
  for (int i = 0; i < 26 && (i + 2) < n_in; ++i) p[i] = (const float*)d_in[i + 2];
  // row: p[0]=ln_g p[1]=ln_b p[2]=Wq p[3]=bq p[4]=Wk p[5]=bk p[6]=Wv p[7]=bv p[8]=Wo p[9]=bo
  // col: p[10..19] same order;  ffn: p[20]=ln_g p[21]=ln_b p[22]=W1 p[23]=b1 p[24]=W2 p[25]=b2

  char* ws = (char*)d_ws;
  size_t off = 0;
  float*          xbuf = (float*)(ws + off);          off += (size_t)M_ * D_ * 4;  // x after row attn
  unsigned short* qt   = (unsigned short*)(ws + off); off += (size_t)M_ * D_ * 2;  // q_t / ctx
  unsigned short* kt   = (unsigned short*)(ws + off); off += (size_t)M_ * D_ * 2;
  unsigned short* vt   = (unsigned short*)(ws + off); off += (size_t)M_ * D_ * 2;
  unsigned short* yb   = (unsigned short*)(ws + off); off += (size_t)M_ * D_ * 2;  // LN output
  float*          sc   = (float*)(ws + off);          off += (size_t)H_ * 256 * 256 * 4;
  unsigned short* pr   = (unsigned short*)(ws + off); off += (size_t)H_ * 256 * 256 * 2;
  unsigned short* wb   = (unsigned short*)(ws + off);
  const size_t WDD = (size_t)D_ * D_, WFD = (size_t)F_ * D_;
  unsigned short* w_rq = wb;              unsigned short* w_rk = w_rq + WDD;
  unsigned short* w_rv = w_rk + WDD;      unsigned short* w_ro = w_rv + WDD;
  unsigned short* w_cq = w_ro + WDD;      unsigned short* w_ck = w_cq + WDD;
  unsigned short* w_cv = w_ck + WDD;      unsigned short* w_co = w_cv + WDD;
  unsigned short* w_f1 = w_co + WDD;      unsigned short* w_f2 = w_f1 + WFD;
  unsigned short* h1   = (unsigned short*)xbuf;  // 402.7MB: aliases xbuf+qt+kt (free at FFN time)

  const dim3 blk(256);
  auto cvt = [&](const float* s, unsigned short* d, size_t n) {
    cvt_bf16<<<dim3((unsigned)((n + 255) / 256)), blk, 0, stream>>>(s, d, (int)n);
  };
  cvt(p[2],  w_rq, WDD); cvt(p[4],  w_rk, WDD); cvt(p[6],  w_rv, WDD); cvt(p[8],  w_ro, WDD);
  cvt(p[12], w_cq, WDD); cvt(p[14], w_ck, WDD); cvt(p[16], w_cv, WDD); cvt(p[18], w_co, WDD);
  cvt(p[22], w_f1, WFD); cvt(p[24], w_f2, WFD);

  const float qscale = 0.0078125f;  // HD^-0.5 / sqrt(256) = (1/8)/16
  const dim3 gProj(D_ / 64, M_ / 128, 1);
  const dim3 gScor(256 / 64, 256 / 128, H_);
  const dim3 gCtx((R_ * HD_) / 64, 256 / 128, H_);

  // ---------------- row attention ----------------
  ln_kernel<<<dim3(M_), blk, 0, stream>>>(x_in, p[0], p[1], yb, 0);
  gemm_wmma<MODE_PROJ_Q><<<gProj, blk, 0, stream>>>(yb, w_rq, p[3], nullptr, nullptr, qt, pm, D_, D_, D_, qscale, 0);
  gemm_wmma<MODE_PROJ_K><<<gProj, blk, 0, stream>>>(yb, w_rk, p[5], nullptr, nullptr, kt, pm, D_, D_, D_, 1.f, 0);
  gemm_wmma<MODE_PROJ_V><<<gProj, blk, 0, stream>>>(yb, w_rv, p[7], nullptr, nullptr, vt, pm, D_, D_, D_, 1.f, 0);
  gemm_wmma<MODE_SCORES><<<gScor, blk, 0, stream>>>(qt, kt, nullptr, nullptr, sc, nullptr, pm, R_ * HD_, R_ * HD_, R_ * HD_, 1.f, 0);
  softmax_kernel<<<dim3(H_ * 256), blk, 0, stream>>>(sc, pr, pm, 0);
  gemm_wmma<MODE_CTX><<<gCtx, blk, 0, stream>>>(pr, vt, nullptr, nullptr, nullptr, qt, pm, 256, 256, 256, 1.f, 0);
  gemm_wmma<MODE_OUTPROJ><<<gProj, blk, 0, stream>>>(qt, w_ro, p[9], x_in, xbuf, nullptr, pm, D_, D_, D_, 1.f, 0);

  // ---------------- column attention (transposed domain) ----------------
  ln_kernel<<<dim3(M_), blk, 0, stream>>>(xbuf, p[10], p[11], yb, 1);
  gemm_wmma<MODE_PROJ_Q><<<gProj, blk, 0, stream>>>(yb, w_cq, p[13], nullptr, nullptr, qt, pm, D_, D_, D_, qscale, 1);
  gemm_wmma<MODE_PROJ_K><<<gProj, blk, 0, stream>>>(yb, w_ck, p[15], nullptr, nullptr, kt, pm, D_, D_, D_, 1.f, 0);
  gemm_wmma<MODE_PROJ_V><<<gProj, blk, 0, stream>>>(yb, w_cv, p[17], nullptr, nullptr, vt, pm, D_, D_, D_, 1.f, 0);
  gemm_wmma<MODE_SCORES><<<gScor, blk, 0, stream>>>(qt, kt, nullptr, nullptr, sc, nullptr, pm, C_ * HD_, C_ * HD_, C_ * HD_, 1.f, 0);
  softmax_kernel<<<dim3(H_ * 256), blk, 0, stream>>>(sc, pr, pm, 1);
  gemm_wmma<MODE_CTX><<<gCtx, blk, 0, stream>>>(pr, vt, nullptr, nullptr, nullptr, qt, pm, 256, 256, 256, 1.f, 0);
  gemm_wmma<MODE_OUTPROJ><<<gProj, blk, 0, stream>>>(qt, w_co, p[19], xbuf, (float*)d_out, nullptr, pm, D_, D_, D_, 1.f, 1);

  // ---------------- FFN ----------------
  ln_kernel<<<dim3(M_), blk, 0, stream>>>((const float*)d_out, p[20], p[21], yb, 0);
  gemm_wmma<MODE_FFN1><<<dim3(F_ / 64, M_ / 128, 1), blk, 0, stream>>>(yb, w_f1, p[23], nullptr, nullptr, h1, pm, D_, D_, D_, 1.f, 0);
  gemm_wmma<MODE_FFN2><<<dim3(D_ / 64, M_ / 128, 1), blk, 0, stream>>>(h1, w_f2, p[25], (const float*)d_out, (float*)d_out, nullptr, pm, F_, F_, F_, 1.f, 0);
}
